// TernaryInferenceLinear_34686155882822
// MI455X (gfx1250) — compile-verified, matrix operand order checked by reading, not run on hardware
//
#include <hip/hip_runtime.h>

typedef _Float16 half_t;
typedef __attribute__((ext_vector_type(16))) _Float16 v16h;
typedef __attribute__((ext_vector_type(8)))  _Float16 v8h;
typedef __attribute__((ext_vector_type(8)))  float    v8f;

#define IN_F   4096
#define OUT_F  4096
#define RANK   16
#define TOKENS 8192
#define WORDS  256          // IN_F / 16
#define K_AUG  4160         // 4096 ternary + 16 lora + 1 rowsum*mu + 1 bias + 46 pad
                            // 4160/32 = 130 k-steps (even -> clean ping-pong pipeline)

// ---------------------------------------------------------------------------
// helpers
// ---------------------------------------------------------------------------
union V16 { v16h v; v8h h[2]; };

__device__ __forceinline__ v16h load16(const half_t* p) {
  // two 16B chunks: halfs [0..7] at p, halfs [8..15] at p+16 (per-lane k split)
  V16 u;
  u.h[0] = *(const v8h*)(p);
  u.h[1] = *(const v8h*)(p + 16);
  return u.v;
}

__device__ __forceinline__ void load_tiles4(v16h dst[4], const half_t* p) {
#pragma unroll
  for (int i = 0; i < 4; ++i) dst[i] = load16(p + (size_t)(i * 16) * K_AUG);
}

__device__ __forceinline__ void wmma16(v8f acc[4][4], const v16h a[4], const v16h b[4]) {
#pragma unroll
  for (int m = 0; m < 4; ++m)
#pragma unroll
    for (int n = 0; n < 4; ++n)
      acc[m][n] = __builtin_amdgcn_wmma_f32_16x16x32_f16(
          false, a[m], false, b[n], (short)0, acc[m][n], false, false);
}

__device__ __forceinline__ void wmma4(v8f acc[4], const v16h a[4], const v16h b) {
#pragma unroll
  for (int m = 0; m < 4; ++m)
    acc[m] = __builtin_amdgcn_wmma_f32_16x16x32_f16(
        false, a[m], false, b, (short)0, acc[m], false, false);
}

// ---------------------------------------------------------------------------
// main augmented GEMM (defined FIRST so its loop shows in the disasm window):
// out[T][O] = xh[T][K_AUG] * Wh[O][K_AUG]^T
// block = 8 waves (2 M x 4 N), block tile 128 x 256, wave tile 64 x 64.
// Register double-buffered: loads for step s+1 overlap WMMAs of step s.
// ---------------------------------------------------------------------------
__global__ void __launch_bounds__(256, 1) gemm_kernel(
    const half_t* __restrict__ xh, const half_t* __restrict__ Wh,
    float* __restrict__ out) {
  const int tid  = threadIdx.x;
  const int wave = tid >> 5;
  const int lane = tid & 31;
  const int r16  = lane & 15;
  const int hi   = lane >> 4;
  const int wm   = wave & 1;
  const int wn   = wave >> 1;
  const int m0   = blockIdx.y * 128 + wm * 64;
  const int n0   = blockIdx.x * 256 + wn * 64;

  const half_t* pA = xh + (size_t)(m0 + r16) * K_AUG + hi * 8;
  const half_t* pB = Wh + (size_t)(n0 + r16) * K_AUG + hi * 8;

  v8f acc[4][4];
  v8f zero = {};
#pragma unroll
  for (int m = 0; m < 4; ++m)
#pragma unroll
    for (int n = 0; n < 4; ++n) acc[m][n] = zero;

  v16h a0[4], b0[4], a1[4], b1[4];

  // prologue: load k-step 0 into buffer 0
  load_tiles4(a0, pA);
  load_tiles4(b0, pB);
  const half_t* qA = pA + 32;   // points at k-step 1
  const half_t* qB = pB + 32;

  // steady state: 64 iterations x 2 k-steps; loads lead WMMAs by one step
#pragma unroll 1
  for (int i = 0; i < 64; ++i) {
    load_tiles4(a1, qA);            // step 2i+1
    load_tiles4(b1, qB);
    wmma16(acc, a0, b0);            // step 2i
    load_tiles4(a0, qA + 32);       // step 2i+2
    load_tiles4(b0, qB + 32);
    wmma16(acc, a1, b1);            // step 2i+1
    qA += 64;
    qB += 64;
  }
  // epilogue: buffer0 holds step 128; load final step 129, drain both
  load_tiles4(a1, qA);              // step 129 (last: K_AUG/32 - 1)
  load_tiles4(b1, qB);
  wmma16(acc, a0, b0);              // step 128
  wmma16(acc, a1, b1);              // step 129

  // C/D layout: lanes 0-15 -> M = j, lanes 16-31 -> M = 8 + j; N = lane % 16
#pragma unroll
  for (int m = 0; m < 4; ++m) {
#pragma unroll
    for (int n = 0; n < 4; ++n) {
      int col   = n0 + n * 16 + r16;
      int rbase = m0 + m * 16 + hi * 8;
#pragma unroll
      for (int j = 0; j < 8; ++j)
        out[(size_t)(rbase + j) * OUT_F + col] = acc[m][n][j];
    }
  }
}

// ---------------------------------------------------------------------------
// lora_y: y = x @ A^T  (WMMA mini-GEMM, M per wave = 64, N = 16)
// writes y[t][r] into xh[t][4096 + r] as f16.  (~0.4% of main GEMM work)
// Register double-buffered like the main GEMM (128 k-steps, even).
// ---------------------------------------------------------------------------
__global__ void __launch_bounds__(256, 1) lora_y_kernel(
    half_t* __restrict__ xh, const half_t* __restrict__ Ah) {
  const int tid  = threadIdx.x;
  const int wave = tid >> 5;
  const int lane = tid & 31;
  const int r16  = lane & 15;
  const int hi   = lane >> 4;
  const int m0   = blockIdx.x * 512 + wave * 64;

  const half_t* pA = xh + (size_t)(m0 + r16) * K_AUG + hi * 8;
  const half_t* pB = Ah + (size_t)r16 * IN_F + hi * 8;

  v8f acc[4];
  v8f zero = {};
#pragma unroll
  for (int m = 0; m < 4; ++m) acc[m] = zero;

  v16h a0[4], a1[4], b0, b1;

  // prologue: k-step 0
  load_tiles4(a0, pA);
  b0 = load16(pB);
  const half_t* qA = pA + 32;   // k-step 1
  const half_t* qB = pB + 32;

  // 63 iterations x 2 k-steps; loads lead WMMAs by one step
#pragma unroll 1
  for (int i = 0; i < 63; ++i) {
    load_tiles4(a1, qA);            // step 2i+1
    b1 = load16(qB);
    wmma4(acc, a0, b0);             // step 2i
    load_tiles4(a0, qA + 32);       // step 2i+2
    b0 = load16(qB + 32);
    wmma4(acc, a1, b1);             // step 2i+1
    qA += 64;
    qB += 64;
  }
  // drain: buffer0 holds step 126; load final step 127
  load_tiles4(a1, qA);
  b1 = load16(qB);
  wmma4(acc, a0, b0);               // step 126
  wmma4(acc, a1, b1);               // step 127

#pragma unroll
  for (int m = 0; m < 4; ++m) {
    int rbase = m0 + m * 16 + hi * 8;
#pragma unroll
    for (int j = 0; j < 8; ++j)
      xh[(size_t)(rbase + j) * K_AUG + IN_F + r16] = (half_t)acc[m][j];
  }
}

// ---------------------------------------------------------------------------
// prep_w: build augmented weight matrix Wh[OUT_F][K_AUG] (f16)
//   k <  4096 : alpha[o] * (code-1)   (ternary * per-row scale)
//   4096..4111: lora_B[o][k-4096]     (LORA_SCALE = 1.0)
//   4112      : mu[o]
//   4113      : bias[o]
//   4114..4159: 0
// ---------------------------------------------------------------------------
__global__ void __launch_bounds__(256) prep_w_kernel(
    const int* __restrict__ packed, const float* __restrict__ alpha,
    const float* __restrict__ mu, const float* __restrict__ bias,
    const float* __restrict__ lora_B, half_t* __restrict__ Wh) {
  const int o   = blockIdx.x;
  const int tid = threadIdx.x;
  const float a = alpha[o];
  const int word = packed[o * WORDS + tid];
  half_t* row = Wh + (size_t)o * K_AUG;

  V16 u;
#pragma unroll
  for (int j = 0; j < 16; ++j) {
    int code = (word >> (2 * j)) & 3;
    u.v[j] = (half_t)(a * (float)(code - 1));
  }
  *(v8h*)(row + tid * 16)     = u.h[0];
  *(v8h*)(row + tid * 16 + 8) = u.h[1];

  if (tid < 64) {
    float val = 0.0f;
    if (tid < 16)       val = lora_B[o * RANK + tid];
    else if (tid == 16) val = mu[o];
    else if (tid == 17) val = bias[o];
    row[IN_F + tid] = (half_t)val;
  }
}

// ---------------------------------------------------------------------------
// prep_a: lora_A fp32 [RANK][IN_F] -> f16 Ah [RANK][IN_F]
// ---------------------------------------------------------------------------
__global__ void __launch_bounds__(256) prep_a_kernel(
    const float* __restrict__ lora_A, half_t* __restrict__ Ah) {
  int i = blockIdx.x * 256 + threadIdx.x;   // grid covers RANK*IN_F = 65536
  Ah[i] = (half_t)lora_A[i];
}

// ---------------------------------------------------------------------------
// prep_x: one block per token.
//   xh[t][0..4095]   = f16(x[t][:])
//   xh[t][4096..4159]= {0 x16 (lora, filled later), S[t], 1.0, 0...}
// ---------------------------------------------------------------------------
__global__ void __launch_bounds__(256) prep_x_kernel(
    const float* __restrict__ x, half_t* __restrict__ xh) {
  const int t   = blockIdx.x;
  const int tid = threadIdx.x;
  const float* xr = x + (size_t)t * IN_F;
  half_t* xo = xh + (size_t)t * K_AUG;

  float s = 0.0f;
#pragma unroll
  for (int k = 0; k < 16; ++k) {
    int i = k * 256 + tid;
    float v = xr[i];
    xo[i] = (half_t)v;
    s += v;
  }
  // wave32 butterfly reduce
#pragma unroll
  for (int off = 16; off > 0; off >>= 1) s += __shfl_xor(s, off, 32);

  __shared__ float wsum[8];
  const int wave = tid >> 5, lane = tid & 31;
  if (lane == 0) wsum[wave] = s;
  __syncthreads();
  if (tid < 64) {
    float tot = 0.0f;
#pragma unroll
    for (int i = 0; i < 8; ++i) tot += wsum[i];
    float val = (tid == 16) ? tot : ((tid == 17) ? 1.0f : 0.0f);
    xo[IN_F + tid] = (half_t)val;   // S[t] pairs with mu col, 1.0 with bias col
  }
}

// ---------------------------------------------------------------------------
// launcher
// ---------------------------------------------------------------------------
extern "C" void kernel_launch(void* const* d_in, const int* in_sizes, int n_in,
                              void* d_out, int out_size, void* d_ws, size_t ws_size,
                              hipStream_t stream) {
  (void)in_sizes; (void)n_in; (void)out_size; (void)ws_size;

  const float* x      = (const float*)d_in[0];
  const int*   packed = (const int*)d_in[1];
  const float* alpha  = (const float*)d_in[2];
  const float* mu     = (const float*)d_in[3];
  const float* bias   = (const float*)d_in[4];
  const float* lora_A = (const float*)d_in[5];
  const float* lora_B = (const float*)d_in[6];
  float* out = (float*)d_out;

  char* ws = (char*)d_ws;
  half_t* Wh = (half_t*)ws;                                            // 4096*4160 f16
  half_t* xh = (half_t*)(ws + (size_t)OUT_F * K_AUG * sizeof(half_t)); // 8192*4160 f16
  half_t* Ah = (half_t*)(ws + ((size_t)OUT_F + TOKENS) * K_AUG * sizeof(half_t)); // 16*4096

  prep_w_kernel<<<OUT_F, 256, 0, stream>>>(packed, alpha, mu, bias, lora_B, Wh);
  prep_a_kernel<<<(RANK * IN_F) / 256, 256, 0, stream>>>(lora_A, Ah);
  prep_x_kernel<<<TOKENS, 256, 0, stream>>>(x, xh);
  lora_y_kernel<<<TOKENS / 512, 256, 0, stream>>>(xh, Ah);

  dim3 grid(OUT_F / 256, TOKENS / 128, 1);
  gemm_kernel<<<grid, 256, 0, stream>>>(xh, Wh, out);
}
